// MOEBlock_2534030705230
// MI455X (gfx1250) — matrix operand order repaired and checked by reading, not compiled, not run on hardware
//
#include <hip/hip_runtime.h>

// ---------------------------------------------------------------------------
// MoE block for MI455X (gfx1250, wave32, WMMA).
// Sparse top-2 dispatch + shared expert as virtual expert 16.
// Both GEMMs run on v_wmma_f32_16x16x32_bf16.
// A (token) tiles staged to LDS via GLOBAL_LOAD_ASYNC_TO_LDS_B128 (ASYNCcnt).
// ---------------------------------------------------------------------------

#define D_DIM 512        // hidden size
#define I_DIM 384        // moe intermediate size
#define N_TOK 8192       // B*S tokens
#define NEXP  16         // routed experts
#define NE    17         // routed + shared

typedef __attribute__((ext_vector_type(16))) __bf16 v16bf;
typedef __attribute__((ext_vector_type(8)))  __bf16 v8bf;
typedef __attribute__((ext_vector_type(8)))  float  v8f;

union FragU { v16bf v; v8bf p[2]; };

// A-matrix fragment (16x32 bf16, MxK). Layout used by this code:
//   lane l: row m = l&15, half h = l>>4
//   VGPR0-7 hold K = 16h .. 16h+15 (contiguous, matching load_frag_b)
__device__ __forceinline__ v16bf load_frag_a(const __bf16* row_k0, int h) {
    FragU f;
    f.p[0] = *(const v8bf*)(row_k0 + 16 * h);
    f.p[1] = *(const v8bf*)(row_k0 + 16 * h + 8);
    return f.v;
}

// B-matrix fragment (32x16 bf16, KxN), weights stored K-contiguous per column.
// Layout used here (same as A): lanes 0-15 hold K=0..15, lanes 16-31 hold K=16..31.
//   lane l: col n = l&15, half h = l>>4; VGPR0-7 hold K = 16h .. 16h+15.
__device__ __forceinline__ v16bf load_frag_b(const __bf16* col_k0, int h) {
    FragU f;
    f.p[0] = *(const v8bf*)(col_k0 + 16 * h);
    f.p[1] = *(const v8bf*)(col_k0 + 16 * h + 8);
    return f.v;
}

// Async global->LDS copy, 16 bytes per lane (VGLOBAL encoding, ASYNCcnt).
// LDS aperture maps addr[31:0] -> LDS byte offset, so the low 32 bits of the
// generic shared pointer are the DS address operand.
__device__ __forceinline__ void async_load_b128(void* lds_ptr, const void* gptr) {
    unsigned lds_off = (unsigned)(unsigned long long)(uintptr_t)lds_ptr;
    asm volatile("global_load_async_to_lds_b128 %0, %1, off"
                 :: "v"(lds_off), "v"(gptr)
                 : "memory");
}

__device__ __forceinline__ void wait_asynccnt0() {
#if defined(__has_builtin) && __has_builtin(__builtin_amdgcn_s_wait_asynccnt)
    __builtin_amdgcn_s_wait_asynccnt(0);
#else
    asm volatile("s_wait_asynccnt 0x0" ::: "memory");
#endif
}

// ---------------------------------------------------------------------------
// init: zero output accumulator, init per-expert counters (expert 16 = shared)
// ---------------------------------------------------------------------------
__global__ void init_kernel(float* __restrict__ y, int* __restrict__ cnt, size_t ny) {
    size_t i = (size_t)blockIdx.x * blockDim.x + threadIdx.x;
    if (i < ny) y[i] = 0.0f;
    if (i < NE) cnt[i] = (i == NEXP) ? N_TOK : 0;
}

// ---------------------------------------------------------------------------
// fp32 -> bf16 conversion of activations
// ---------------------------------------------------------------------------
__global__ void cvt_x_kernel(const float* __restrict__ x, __bf16* __restrict__ xbf, size_t n) {
    size_t i = (size_t)blockIdx.x * blockDim.x + threadIdx.x;
    if (i < n) xbf[i] = (__bf16)x[i];
}

// W1: (D, 2I) interleaved gate/up -> two transposed bf16 mats [(e*I + c)*D + d]
__global__ void cvt_w1_kernel(const float* __restrict__ ew1, const float* __restrict__ sw1,
                              __bf16* __restrict__ w1gT, __bf16* __restrict__ w1uT) {
    int i = blockIdx.x * blockDim.x + threadIdx.x;
    if (i >= NE * I_DIM * D_DIM) return;
    int e   = i / (I_DIM * D_DIM);
    int rem = i % (I_DIM * D_DIM);
    int c   = rem / D_DIM;
    int d   = rem % D_DIM;
    const float* src = (e < NEXP) ? (ew1 + (size_t)e * D_DIM * 2 * I_DIM) : sw1;
    w1gT[i] = (__bf16)src[(size_t)d * (2 * I_DIM) + 2 * c];
    w1uT[i] = (__bf16)src[(size_t)d * (2 * I_DIM) + 2 * c + 1];
}

// W2: (I, D) -> transposed bf16 [(e*D + d)*I + i]
__global__ void cvt_w2_kernel(const float* __restrict__ ew2, const float* __restrict__ sw2,
                              __bf16* __restrict__ w2T) {
    int i = blockIdx.x * blockDim.x + threadIdx.x;
    if (i >= NE * D_DIM * I_DIM) return;
    int e   = i / (D_DIM * I_DIM);
    int rem = i % (D_DIM * I_DIM);
    int d   = rem / I_DIM;
    int ii  = rem % I_DIM;
    const float* src = (e < NEXP) ? (ew2 + (size_t)e * I_DIM * D_DIM) : sw2;
    w2T[i] = (__bf16)src[(size_t)ii * D_DIM + d];
}

// b1: (2I) interleaved -> bg, bu (fp32, added pre-activation)
__global__ void cvt_b1_kernel(const float* __restrict__ eb1, const float* __restrict__ sb1,
                              float* __restrict__ bg, float* __restrict__ bu) {
    int i = blockIdx.x * blockDim.x + threadIdx.x;
    if (i >= NE * I_DIM) return;
    int e = i / I_DIM;
    int c = i % I_DIM;
    const float* src = (e < NEXP) ? (eb1 + (size_t)e * 2 * I_DIM) : sb1;
    bg[i] = src[2 * c];
    bu[i] = src[2 * c + 1];
}

// ---------------------------------------------------------------------------
// gating: logits = x @ gate_w + b; top-2; renorm among the 2 (softmax cancels)
// scatter token -> per-expert lists with atomics
// ---------------------------------------------------------------------------
__global__ void gate_kernel(const float* __restrict__ x, const float* __restrict__ gw,
                            const float* __restrict__ gb, int* __restrict__ cnt,
                            int* __restrict__ list, float* __restrict__ wlist) {
    int t = blockIdx.x * blockDim.x + threadIdx.x;
    if (t >= N_TOK) return;
    float acc[NEXP];
#pragma unroll
    for (int e = 0; e < NEXP; ++e) acc[e] = gb[e];
    const float* xr = x + (size_t)t * D_DIM;
    for (int d = 0; d < D_DIM; ++d) {
        float xv = xr[d];
        const float* g = gw + d * NEXP;
#pragma unroll
        for (int e = 0; e < NEXP; ++e) acc[e] = fmaf(xv, g[e], acc[e]);
    }
    int i0 = 0; float l0 = acc[0];
#pragma unroll
    for (int e = 1; e < NEXP; ++e) if (acc[e] > l0) { l0 = acc[e]; i0 = e; }
    int i1 = (i0 == 0) ? 1 : 0; float l1 = acc[i1];
#pragma unroll
    for (int e = 0; e < NEXP; ++e)
        if (e != i0 && acc[e] > l1) { l1 = acc[e]; i1 = e; }
    // top-2 of softmax, renormalized: w0 = e^l0/(e^l0+e^l1)
    float w0 = 1.0f / (1.0f + __expf(l1 - l0));
    float w1 = 1.0f - w0;
    int p0 = atomicAdd(&cnt[i0], 1);
    list[i0 * N_TOK + p0]  = t;
    wlist[i0 * N_TOK + p0] = w0;
    int p1 = atomicAdd(&cnt[i1], 1);
    list[i1 * N_TOK + p1]  = t;
    wlist[i1 * N_TOK + p1] = w1;
    // shared expert: identity list, weight 1
    list[NEXP * N_TOK + t]  = t;
    wlist[NEXP * N_TOK + t] = 1.0f;
}

// ---------------------------------------------------------------------------
// Fused expert MLP: per block = 16 tokens of one expert, 8 waves.
// Stage 0: gather the 16 token rows of x (16 KB) into LDS once per block
//          with async global->LDS b128 copies (ASYNCcnt).
// Phase 1: G/U = X@W1 (WMMA bf16, A from LDS), SwiGLU5 -> t in LDS.
// Phase 2: out = t@W2 (WMMA bf16, A from LDS), scale by routing weight,
//          atomic-add into y.
// ---------------------------------------------------------------------------
__global__ __launch_bounds__(256)
void moe_expert_kernel(const __bf16* __restrict__ xbf,
                       const __bf16* __restrict__ w1gT,
                       const __bf16* __restrict__ w1uT,
                       const __bf16* __restrict__ w2T,
                       const float* __restrict__ bg,
                       const float* __restrict__ bu,
                       const float* __restrict__ sb2,
                       const float* __restrict__ eb2,
                       const float* __restrict__ sact,
                       const float* __restrict__ eact,
                       const int* __restrict__ cnt,
                       const int* __restrict__ list,
                       const float* __restrict__ wlist,
                       float* __restrict__ y) {
    const int e   = blockIdx.y;
    const int m_t = blockIdx.x;
    const int c   = cnt[e];
    if (m_t * 16 >= c) return;     // early-exit for tiles past this expert's load

    __shared__ __bf16 x_lds[16][D_DIM];   // 16 KB  (A tile for phase 1)
    __shared__ __bf16 t_lds[16][I_DIM];   // 12 KB  (A tile for phase 2)
    __shared__ int    s_tok[16];
    __shared__ float  s_wt[16];

    const int tid  = threadIdx.x;
    const int lane = tid & 31;
    const int wv   = tid >> 5;     // 8 waves
    const int n    = lane & 15;
    const int h    = lane >> 4;

    if (tid < 16) {
        int slot = m_t * 16 + tid;
        if (slot < c) {
            s_tok[tid] = list[e * N_TOK + slot];
            s_wt[tid]  = wlist[e * N_TOK + slot];
        } else {
            s_tok[tid] = 0;        // pad rows read token 0, weight 0 => no-op
            s_wt[tid]  = 0.0f;
        }
    }
    __syncthreads();

    // ---- Stage 0: async gather of A tile into LDS --------------------------
    // 16 rows x 1024 B = 64 x b128 per row; 256 lanes x 4 chunks.
#pragma unroll
    for (int it = 0; it < 4; ++it) {
        int i     = tid + it * 256;       // 0 .. 1023
        int m     = i >> 6;               // token row
        int chunk = i & 63;               // 16-byte chunk within row
        const __bf16* g = xbf + (size_t)s_tok[m] * D_DIM + chunk * 8;
        async_load_b128(&x_lds[m][chunk * 8], g);
    }
    wait_asynccnt0();
    __syncthreads();

    // 5 learned activation scalars (clamps via softplus)
    const float* act = (e < NEXP) ? (eact + e * 5) : sact;
    const float alpha  = act[0];
    const float gscale = act[1];
    const float ushift = act[2];
    const float gclamp = logf(1.0f + __expf(act[3]));
    const float uclamp = logf(1.0f + __expf(act[4]));

    const __bf16* arow = &x_lds[n][0];   // this lane's A row (LDS)

    // ---- Phase 1: 24 paired (G,U) column tiles, 3 per wave -----------------
    for (int ct = wv * 3; ct < wv * 3 + 3; ++ct) {
        v8f gacc = {};
        v8f uacc = {};
        const int col = ct * 16 + n;
        const __bf16* bgrow = w1gT + (size_t)(e * I_DIM + col) * D_DIM;
        const __bf16* burow = w1uT + (size_t)(e * I_DIM + col) * D_DIM;
#pragma unroll
        for (int k0 = 0; k0 < D_DIM; k0 += 32) {
            v16bf a  = load_frag_a(arow + k0, h);   // ds_load_b128 x2
            v16bf b0 = load_frag_b(bgrow + k0, h);  // global_load_b128 x2
            v16bf b1 = load_frag_b(burow + k0, h);
            gacc = __builtin_amdgcn_wmma_f32_16x16x32_bf16(
                       false, a, false, b0, (short)0, gacc, false, false);
            uacc = __builtin_amdgcn_wmma_f32_16x16x32_bf16(
                       false, a, false, b1, (short)0, uacc, false, false);
        }
        const float bgv = bg[e * I_DIM + col];
        const float buv = bu[e * I_DIM + col];
#pragma unroll
        for (int r = 0; r < 8; ++r) {
            float g = gacc[r] + bgv;
            float u = uacc[r] + buv;
            g = fminf(fmaxf(g, -gclamp), gclamp);
            u = fminf(fmaxf(u, -uclamp), uclamp);
            float sig = 1.0f / (1.0f + __expf(-g * alpha));
            float tv  = g * sig * gscale * (u + ushift);
            t_lds[r + 8 * h][col] = (__bf16)tv;   // C layout: m = r + 8h, n = lane&15
        }
    }
    __syncthreads();

    // ---- Phase 2: 32 output D tiles, 4 per wave ----------------------------
    const float* b2p = (e < NEXP) ? (eb2 + e * D_DIM) : sb2;
    const __bf16* alds = &t_lds[n][0];            // lane's A row in LDS
    for (int dt = wv * 4; dt < wv * 4 + 4; ++dt) {
        v8f acc = {};
        const int d = dt * 16 + n;
        const __bf16* brow = w2T + (size_t)(e * D_DIM + d) * I_DIM;
#pragma unroll
        for (int k0 = 0; k0 < I_DIM; k0 += 32) {
            v16bf a = load_frag_a(alds + k0, h);  // ds_load_b128 x2
            v16bf b = load_frag_b(brow + k0, h);
            acc = __builtin_amdgcn_wmma_f32_16x16x32_bf16(
                      false, a, false, b, (short)0, acc, false, false);
        }
        const float b2v = b2p[d];
#pragma unroll
        for (int r = 0; r < 8; ++r) {
            int m = r + 8 * h;
            float val = s_wt[m] * (acc[r] + b2v);
            atomicAdd(&y[(size_t)s_tok[m] * D_DIM + d], val);
        }
    }
}

// ---------------------------------------------------------------------------
// host-side launcher
// ---------------------------------------------------------------------------
extern "C" void kernel_launch(void* const* d_in, const int* in_sizes, int n_in,
                              void* d_out, int out_size, void* d_ws, size_t ws_size,
                              hipStream_t stream) {
    const float* x    = (const float*)d_in[0];
    const float* gw   = (const float*)d_in[1];
    const float* gb   = (const float*)d_in[2];
    const float* sw1  = (const float*)d_in[3];
    const float* sb1  = (const float*)d_in[4];
    const float* sw2  = (const float*)d_in[5];
    const float* sb2  = (const float*)d_in[6];
    const float* sact = (const float*)d_in[7];
    const float* ew1  = (const float*)d_in[8];
    const float* eb1  = (const float*)d_in[9];
    const float* ew2  = (const float*)d_in[10];
    const float* eb2  = (const float*)d_in[11];
    const float* eact = (const float*)d_in[12];
    float* y = (float*)d_out;

    // workspace carve-out (~30 MB)
    char* ws = (char*)d_ws;
    size_t off = 0;
    auto take = [&](size_t bytes) -> char* {
        char* p = ws + off;
        off = (off + bytes + 255) & ~(size_t)255;
        return p;
    };
    __bf16* xbf  = (__bf16*)take((size_t)N_TOK * D_DIM * sizeof(__bf16));
    __bf16* w1gT = (__bf16*)take((size_t)NE * I_DIM * D_DIM * sizeof(__bf16));
    __bf16* w1uT = (__bf16*)take((size_t)NE * I_DIM * D_DIM * sizeof(__bf16));
    __bf16* w2T  = (__bf16*)take((size_t)NE * D_DIM * I_DIM * sizeof(__bf16));
    float*  bgp  = (float*)take((size_t)NE * I_DIM * sizeof(float));
    float*  bup  = (float*)take((size_t)NE * I_DIM * sizeof(float));
    int*    cnt  = (int*)take(NE * sizeof(int));
    int*    list = (int*)take((size_t)NE * N_TOK * sizeof(int));
    float*  wls  = (float*)take((size_t)NE * N_TOK * sizeof(float));

    const int TB = 256;
    const size_t ny = (size_t)N_TOK * D_DIM;

    init_kernel<<<(unsigned)((ny + TB - 1) / TB), TB, 0, stream>>>(y, cnt, ny);
    cvt_x_kernel<<<(unsigned)((ny + TB - 1) / TB), TB, 0, stream>>>(x, xbf, ny);
    {
        int tot = NE * I_DIM * D_DIM;
        cvt_w1_kernel<<<(tot + TB - 1) / TB, TB, 0, stream>>>(ew1, sw1, w1gT, w1uT);
    }
    {
        int tot = NE * D_DIM * I_DIM;
        cvt_w2_kernel<<<(tot + TB - 1) / TB, TB, 0, stream>>>(ew2, sw2, w2T);
    }
    {
        int tot = NE * I_DIM;
        cvt_b1_kernel<<<(tot + TB - 1) / TB, TB, 0, stream>>>(eb1, sb1, bgp, bup);
    }
    gate_kernel<<<N_TOK / TB, TB, 0, stream>>>(x, gw, gb, cnt, list, wls);

    dim3 grid(N_TOK / 16, NE);   // token tiles x (routed experts + shared)
    moe_expert_kernel<<<grid, TB, 0, stream>>>(xbf, w1gT, w1uT, w2T, bgp, bup,
                                               sb2, eb2, sact, eact,
                                               cnt, list, wls, y);
}
